// MultiHeadSelfAttention_5188320494351
// MI455X (gfx1250) — matrix operand order repaired and checked by reading, not compiled, not run on hardware
//
#include <hip/hip_runtime.h>
#include <hip/hip_bf16.h>

// ---------------------------------------------------------------------------
// Multi-head self-attention, forward.  All GEMMs via v_wmma_f32_16x16x32_bf16.
//   B=2, T=2048, C=1024, H=16, D=64
// Pipeline (all on `stream`, serialized by stream order):
//   1) pack x, qkv_w, out_w  fp32 -> bf16
//   2) QKV GEMM (bf16 WMMA, fp32 accum) -> Q(scaled), K  [B,H,T,D]; V^T [B,H,D,T]
//   3) flash attention per 16-query tile: S^T = K Q^T (WMMA), online softmax,
//      O^T = V^T P^T (WMMA)  -> O bf16 [B,T,C]
//   4) out projection GEMM (bf16 WMMA) + bias -> fp32 d_out
// ---------------------------------------------------------------------------

typedef __attribute__((ext_vector_type(16))) __bf16 v16bf;
typedef __attribute__((ext_vector_type(8)))  float  v8f;

constexpr int BATCH = 2;
constexpr int T  = 2048;
constexpr int C  = 1024;
constexpr int H  = 16;
constexpr int D  = 64;
constexpr int BT = BATCH * T;   // 4096
constexpr int C3 = 3 * C;       // 3072

static __device__ __forceinline__ __bf16 f2bf(float f) {
  unsigned u = __builtin_bit_cast(unsigned, f);
  unsigned r = u + 0x7FFFu + ((u >> 16) & 1u);   // round to nearest even
  unsigned short h = (unsigned short)(r >> 16);
  return __builtin_bit_cast(__bf16, h);
}

static __device__ __forceinline__ v8f wmma_bf16(v16bf a, v16bf b, v8f c) {
  // (neg_a, A, neg_b, B, c_mod, C, reuse_a, reuse_b)
  return __builtin_amdgcn_wmma_f32_16x16x32_bf16(false, a, false, b, (short)0, c,
                                                 false, false);
}

// A-operand fragment (16x32 bf16, row-major source; row = row0 + (lane&15)).
// Lanes 0-15 hold K k0..k0+15; lanes 16-31 hold K k0+16..k0+31.
static __device__ __forceinline__ v16bf load_frag_a(const __bf16* __restrict__ p,
                                                    int ld, int row0, int k0) {
  const int lane = threadIdx.x & 31;
  const int half = lane >> 4, r = lane & 15;
  const __bf16* q = p + (size_t)(row0 + r) * ld + k0 + half * 16;
  v16bf f;
#pragma unroll
  for (int j = 0; j < 16; ++j) f[j] = q[j];
  return f;
}

// B-operand fragment (32x16 bf16) from a source where B column n is a
// contiguous-k row (row = col0 + (lane&15)).  Lanes 0-15 hold K k0..k0+15,
// lanes 16-31 hold K k0+16..k0+31.
static __device__ __forceinline__ v16bf load_frag_b(const __bf16* __restrict__ p,
                                                    int ld, int col0, int k0) {
  const int lane = threadIdx.x & 31;
  const int half = lane >> 4, n = lane & 15;
  const __bf16* q = p + (size_t)(col0 + n) * ld + k0 + half * 16;
  v16bf f;
#pragma unroll
  for (int j = 0; j < 16; ++j) f[j] = q[j];
  return f;
}

// ---------------------------------------------------------------------------
__global__ void cvt_f32_bf16(const float* __restrict__ s, __bf16* __restrict__ d, int n) {
  int i = blockIdx.x * blockDim.x + threadIdx.x;
  const int st = gridDim.x * blockDim.x;
  for (; i < n; i += st) d[i] = f2bf(s[i]);
}

// ---------------------------------------------------------------------------
// QKV GEMM: [BT,C] @ qkv_w^T -> split/scatter into Q,K (B,H,T,D) and V^T (B,H,D,T)
// Wave tile: 32 (M) x 64 (N).  Waves = (BT/32)*(C3/64) = 128*48 = 6144.
__global__ void qkv_gemm_kernel(const __bf16* __restrict__ xb,
                                const __bf16* __restrict__ w,
                                const float*  __restrict__ bias,
                                __bf16* __restrict__ Qb,
                                __bf16* __restrict__ Kb,
                                __bf16* __restrict__ Vt) {
  const int lane = threadIdx.x & 31;
  const int wave = blockIdx.x * (blockDim.x >> 5) + (threadIdx.x >> 5);
  const int mTiles = BT / 32;                 // 128
  const int mTile = wave % mTiles;
  const int nTile = wave / mTiles;            // 0..47
  const int m0 = mTile * 32, n0 = nTile * 64;

  v8f acc[2][4] = {};
  for (int k0 = 0; k0 < C; k0 += 32) {
    v16bf a0 = load_frag_a(xb, C, m0,      k0);
    v16bf a1 = load_frag_a(xb, C, m0 + 16, k0);
#pragma unroll
    for (int nf = 0; nf < 4; ++nf) {
      v16bf b = load_frag_b(w, C, n0 + nf * 16, k0);
      acc[0][nf] = wmma_bf16(a0, b, acc[0][nf]);
      acc[1][nf] = wmma_bf16(a1, b, acc[1][nf]);
    }
  }

  const int half = lane >> 4, nl = lane & 15;
#pragma unroll
  for (int mi = 0; mi < 2; ++mi) {
#pragma unroll
    for (int nf = 0; nf < 4; ++nf) {
      const int n = n0 + nf * 16 + nl;
      const float bv = bias[n];
      const int which = n >> 10;              // 0:Q 1:K 2:V
      const int c = n & (C - 1);
      const int h = c >> 6, d = c & (D - 1);
#pragma unroll
      for (int v = 0; v < 8; ++v) {
        const int m = m0 + mi * 16 + half * 8 + v;
        const int bb = m >> 11, t = m & (T - 1);
        const float val = acc[mi][nf][v] + bv;
        if (which == 0) {
          Qb[((size_t)(bb * H + h) * T + t) * D + d] = f2bf(val * 0.125f); // 1/sqrt(64)
        } else if (which == 1) {
          Kb[((size_t)(bb * H + h) * T + t) * D + d] = f2bf(val);
        } else {
          Vt[((size_t)(bb * H + h) * D + d) * T + t] = f2bf(val);
        }
      }
    }
  }
}

// ---------------------------------------------------------------------------
// Flash attention: each wave owns 16 queries x D=64 for one (b,h).
// Waves = B*H * T/16 = 32*128 = 4096.
__global__ void attn_kernel(const __bf16* __restrict__ Qb,
                            const __bf16* __restrict__ Kb,
                            const __bf16* __restrict__ Vt,
                            __bf16* __restrict__ Ob) {
  const int lane = threadIdx.x & 31;
  const int wave = blockIdx.x * (blockDim.x >> 5) + (threadIdx.x >> 5);
  const int qTiles = T / 16;                  // 128
  const int bh = wave / qTiles;
  const int q0 = (wave % qTiles) * 16;

  const __bf16* Qp = Qb + (size_t)bh * T * D;
  const __bf16* Kp = Kb + (size_t)bh * T * D;
  const __bf16* Vp = Vt + (size_t)bh * D * T;

  // Q as B-operand of S^T = K * Q^T  (column t contiguous in d) — loop invariant.
  v16bf qB0 = load_frag_b(Qp, D, q0, 0);
  v16bf qB1 = load_frag_b(Qp, D, q0, 32);

  v8f o[4] = {};                              // O^T accum: [d rows][t cols]
  float run_max = -__builtin_inff();
  float run_sum = 0.0f;
  const int half = lane >> 4;

  for (int s0 = 0; s0 < T; s0 += 32) {
    // S^T frags for key blocks [s0,s0+16) and [s0+16,s0+32)
    v8f st0 = {}, st1 = {};
    {
      v16bf ka = load_frag_a(Kp, D, s0, 0);
      v16bf kb = load_frag_a(Kp, D, s0, 32);
      st0 = wmma_bf16(ka, qB0, st0);
      st0 = wmma_bf16(kb, qB1, st0);
    }
    {
      v16bf ka = load_frag_a(Kp, D, s0 + 16, 0);
      v16bf kb = load_frag_a(Kp, D, s0 + 16, 32);
      st1 = wmma_bf16(ka, qB0, st1);
      st1 = wmma_bf16(kb, qB1, st1);
    }

    // Online softmax over key axis.  Lane (t) and lane^16 (same t) split s.
    float tmax = -__builtin_inff();
#pragma unroll
    for (int v = 0; v < 8; ++v) {
      tmax = fmaxf(tmax, st0[v]);
      tmax = fmaxf(tmax, st1[v]);
    }
    tmax = fmaxf(tmax, __shfl_xor(tmax, 16, 32));
    const float nmax  = fmaxf(run_max, tmax);
    const float scale = __expf(run_max - nmax);

    float p0[8], p1[8];
    float tsum = 0.0f;
#pragma unroll
    for (int v = 0; v < 8; ++v) {
      p0[v] = __expf(st0[v] - nmax);
      p1[v] = __expf(st1[v] - nmax);
      tsum += p0[v] + p1[v];
    }
    tsum += __shfl_xor(tsum, 16, 32);
    run_sum = run_sum * scale + tsum;
    run_max = nmax;

    // Rescale previous accumulation.
#pragma unroll
    for (int di = 0; di < 4; ++di) o[di] = o[di] * scale;

    // Build P^T B-operand from the S^T accumulators (half-wave exchange only):
    //   lanes 0-15 need k=s_local 0..15  -> own st0 + partner st0
    //   lanes 16-31 need k=s_local 16..31 -> partner st1 + own st1
    float x0[8], x1[8];
#pragma unroll
    for (int v = 0; v < 8; ++v) {
      x0[v] = __shfl_xor(p0[v], 16, 32);
      x1[v] = __shfl_xor(p1[v], 16, 32);
    }
    v16bf pB;
#pragma unroll
    for (int v = 0; v < 8; ++v) {
      pB[v]     = f2bf(half ? x1[v] : p0[v]);
      pB[8 + v] = f2bf(half ? p1[v] : x0[v]);
    }

    // O^T += V^T * P^T
#pragma unroll
    for (int di = 0; di < 4; ++di) {
      v16bf va = load_frag_a(Vp, T, di * 16, s0);
      o[di] = wmma_bf16(va, pB, o[di]);
    }
  }

  // Normalize and store O as [B,T,C] bf16 (contiguous c = h*D+d).
  const float inv = 1.0f / run_sum;
  const int nl = lane & 15;
  const int t  = q0 + nl;
  const int bb = bh >> 4, h = bh & (H - 1);
#pragma unroll
  for (int di = 0; di < 4; ++di) {
#pragma unroll
    for (int v = 0; v < 8; ++v) {
      const int d = di * 16 + half * 8 + v;
      Ob[((size_t)(bb * T + t)) * C + h * D + d] = f2bf(o[di][v] * inv);
    }
  }
}

// ---------------------------------------------------------------------------
// Output projection: [BT,C] @ out_w^T + out_b -> fp32 d_out.
// Wave tile 32x64.  Waves = (BT/32)*(C/64) = 128*16 = 2048.
__global__ void out_gemm_kernel(const __bf16* __restrict__ Ob,
                                const __bf16* __restrict__ w,
                                const float*  __restrict__ bias,
                                float* __restrict__ out) {
  const int lane = threadIdx.x & 31;
  const int wave = blockIdx.x * (blockDim.x >> 5) + (threadIdx.x >> 5);
  const int mTiles = BT / 32;                 // 128
  const int mTile = wave % mTiles;
  const int nTile = wave / mTiles;            // 0..15
  const int m0 = mTile * 32, n0 = nTile * 64;

  v8f acc[2][4] = {};
  for (int k0 = 0; k0 < C; k0 += 32) {
    v16bf a0 = load_frag_a(Ob, C, m0,      k0);
    v16bf a1 = load_frag_a(Ob, C, m0 + 16, k0);
#pragma unroll
    for (int nf = 0; nf < 4; ++nf) {
      v16bf b = load_frag_b(w, C, n0 + nf * 16, k0);
      acc[0][nf] = wmma_bf16(a0, b, acc[0][nf]);
      acc[1][nf] = wmma_bf16(a1, b, acc[1][nf]);
    }
  }

  const int half = lane >> 4, nl = lane & 15;
#pragma unroll
  for (int mi = 0; mi < 2; ++mi) {
#pragma unroll
    for (int nf = 0; nf < 4; ++nf) {
      const int n = n0 + nf * 16 + nl;
      const float bv = bias[n];
#pragma unroll
      for (int v = 0; v < 8; ++v) {
        const int m = m0 + mi * 16 + half * 8 + v;
        out[(size_t)m * C + n] = acc[mi][nf][v] + bv;
      }
    }
  }
}

// ---------------------------------------------------------------------------
extern "C" void kernel_launch(void* const* d_in, const int* in_sizes, int n_in,
                              void* d_out, int out_size, void* d_ws, size_t ws_size,
                              hipStream_t stream) {
  const float* x      = (const float*)d_in[0];   // [2,2048,1024]
  const float* qkv_w  = (const float*)d_in[1];   // [3072,1024]
  const float* qkv_b  = (const float*)d_in[2];   // [3072]
  const float* out_w  = (const float*)d_in[3];   // [1024,1024]
  const float* out_b  = (const float*)d_in[4];   // [1024]
  float* out = (float*)d_out;                    // [2,2048,1024]

  // Workspace layout (~48 MB total)
  unsigned char* ws = (unsigned char*)d_ws;
  __bf16* xb   = (__bf16*)ws;  ws += (size_t)BT * C  * sizeof(__bf16); // 8 MB
  __bf16* wqkv = (__bf16*)ws;  ws += (size_t)C3 * C  * sizeof(__bf16); // 6 MB
  __bf16* wo   = (__bf16*)ws;  ws += (size_t)C  * C  * sizeof(__bf16); // 2 MB
  __bf16* Qb   = (__bf16*)ws;  ws += (size_t)BT * C  * sizeof(__bf16); // 8 MB
  __bf16* Kb   = (__bf16*)ws;  ws += (size_t)BT * C  * sizeof(__bf16); // 8 MB
  __bf16* Vt   = (__bf16*)ws;  ws += (size_t)BT * C  * sizeof(__bf16); // 8 MB
  __bf16* Ob   = (__bf16*)ws;  ws += (size_t)BT * C  * sizeof(__bf16); // 8 MB

  cvt_f32_bf16<<<2048, 256, 0, stream>>>(x,     xb,   BT * C);
  cvt_f32_bf16<<<2048, 256, 0, stream>>>(qkv_w, wqkv, C3 * C);
  cvt_f32_bf16<<<1024, 256, 0, stream>>>(out_w, wo,   C * C);

  // 6144 waves, 8 waves (256 threads) per block
  qkv_gemm_kernel<<<768, 256, 0, stream>>>(xb, wqkv, qkv_b, Qb, Kb, Vt);
  // 4096 waves
  attn_kernel<<<512, 256, 0, stream>>>(Qb, Kb, Vt, Ob);
  // 2048 waves
  out_gemm_kernel<<<256, 256, 0, stream>>>(Ob, wo, out_b, out);
}
